// WindowedAttentionLayer_82386062672328
// MI455X (gfx1250) — compile-verified, matrix operand order checked by reading, not run on hardware
//
#include <hip/hip_runtime.h>
#include <hip/hip_bf16.h>

typedef __bf16 bf16_t;
typedef __attribute__((ext_vector_type(16))) __bf16 v16bf;
typedef __attribute__((ext_vector_type(8)))  __bf16 bf16x8;
typedef __attribute__((ext_vector_type(8)))  float  v8f;

constexpr int kD    = 256;
constexpr int kL    = 8192;
constexpr int kB    = 4;
constexpr int kRows = kB * kL;      // 32768
constexpr int kH    = 8;
constexpr int kHD   = 32;
constexpr int kWN   = 9;
constexpr int kP    = 4;

// ---------------- WMMA helpers (gfx1250, wave32) ----------------

__device__ __forceinline__ v8f wmma_bf16(v16bf a, v16bf b, v8f c) {
  // D = A(16x32 bf16) * B(32x16 bf16) + C(16x16 f32)
  return __builtin_amdgcn_wmma_f32_16x16x32_bf16(
      /*neg_a=*/false, a, /*neg_b=*/false, b,
      /*c_mod=*/(short)0, c, /*reuse_a=*/false, /*reuse_b=*/false);
}

// A-operand layout (16-bit, 16x32): lane m(0..15) holds K = {0..7,16..23},
// lane m+16 holds K = {8..15,24..31}. B-operand is symmetric when the
// weight is stored transposed (row = output column, contiguous over K).
__device__ __forceinline__ v16bf load_bf16_tile(const bf16_t* __restrict__ base,
                                                int row0, int k0, int lane) {
  int m  = lane & 15;
  int ko = (lane >> 4) << 3;
  const bf16_t* p = base + (size_t)(row0 + m) * kD + k0 + ko;
  bf16x8 lo = *(const bf16x8*)(p);
  bf16x8 hi = *(const bf16x8*)(p + 16);
  return __builtin_shufflevector(lo, hi, 0,1,2,3,4,5,6,7,8,9,10,11,12,13,14,15);
}

__device__ __forceinline__ v16bf load_f32_tile_cvt(const float* __restrict__ base,
                                                   int row0, int k0, int lane) {
  int m  = lane & 15;
  int ko = (lane >> 4) << 3;
  const float* p = base + (size_t)(row0 + m) * kD + k0 + ko;
  v16bf r;
#pragma unroll
  for (int i = 0; i < 8; ++i) {
    r[i]     = (bf16_t)p[i];
    r[i + 8] = (bf16_t)p[i + 16];
  }
  return r;
}

// ---------------- prep kernels ----------------

__global__ void cvt_x_kernel(const float* __restrict__ x, bf16_t* __restrict__ xbf, int n) {
  int i = blockIdx.x * blockDim.x + threadIdx.x;
  if (i < n) xbf[i] = (bf16_t)x[i];
}

// WB layout: mats 0..3 = Wq^T, Wk^T, Wv^T, Wo^T (bf16); mat 4 = W_out (bf16, untransposed,
// because z = a @ W_out^T means B-operand column j == row j of W_out).
__global__ void prep_w_kernel(const float* __restrict__ Wq, const float* __restrict__ Wk,
                              const float* __restrict__ Wv, const float* __restrict__ Wo,
                              const float* __restrict__ Wout, bf16_t* __restrict__ WB) {
  int i = blockIdx.x * blockDim.x + threadIdx.x;   // 5*65536 total
  int mat = i >> 16;
  int r   = (i >> 8) & 255;
  int c   = i & 255;
  const float* src = (mat == 0) ? Wq : (mat == 1) ? Wk : (mat == 2) ? Wv
                   : (mat == 3) ? Wo : Wout;
  float v = (mat == 4) ? src[r * kD + c] : src[c * kD + r];
  WB[i] = (bf16_t)v;
}

// Ppos[mat][w][c] = (pos @ W)[w][c], fp32. grid = 3*9 blocks of 256.
__global__ __launch_bounds__(256) void posproj_kernel(
    const float* __restrict__ pos, const float* __restrict__ Wq,
    const float* __restrict__ Wk,  const float* __restrict__ Wv,
    float* __restrict__ Ppos) {
  int mat = blockIdx.x / kWN;
  int w   = blockIdx.x % kWN;
  int c   = threadIdx.x;
  const float* W = (mat == 0) ? Wq : (mat == 1) ? Wk : Wv;
  float s = 0.f;
  for (int d = 0; d < kD; ++d) s = fmaf(pos[w * kD + d], W[d * kD + c], s);
  Ppos[((size_t)mat * kWN + w) * kD + c] = s;
}

// ---------------- QKV projection GEMM: Q|K|V = x @ {Wq,Wk,Wv} ----------------
// grid = kRows/16 blocks, 128 threads (4 waves); wave w handles cols [w*64, w*64+64).
__global__ __launch_bounds__(128) void qkv_gemm_kernel(
    const bf16_t* __restrict__ xbf, const bf16_t* __restrict__ WT,
    float* __restrict__ QKV) {
  const int lane = threadIdx.x & 31;
  const int wave = threadIdx.x >> 5;
  const int row0 = blockIdx.x * 16;

  for (int mat = 0; mat < 3; ++mat) {
    const bf16_t* W = WT + (size_t)mat * kD * kD;
    float* Out = QKV + (size_t)mat * kRows * kD;
    v8f acc[4] = {};
#pragma unroll
    for (int kk = 0; kk < 8; ++kk) {
      int k0 = kk * 32;
      v16bf a = load_bf16_tile(xbf, row0, k0, lane);
#pragma unroll
      for (int t = 0; t < 4; ++t) {
        v16bf b = load_bf16_tile(W, wave * 64 + t * 16, k0, lane);
        acc[t] = wmma_bf16(a, b, acc[t]);
      }
    }
    const int n  = lane & 15;
    const int mo = (lane >> 4) * 8;
#pragma unroll
    for (int t = 0; t < 4; ++t) {
      int col0 = wave * 64 + t * 16;
#pragma unroll
      for (int r = 0; r < 8; ++r)
        Out[(size_t)(row0 + mo + r) * kD + col0 + n] = acc[t][r];
    }
  }
}

// ---------------- attention per window ----------------
// ctx[h,d] = sum_k (sum_q softmax_k(scores)[h,q,k]) * v[k,h,d]
__global__ __launch_bounds__(256) void attn_kernel(
    const float* __restrict__ QKV, const float* __restrict__ Ppos,
    float* __restrict__ ctx) {
  __shared__ float qs[kWN][kD];
  __shared__ float ks[kWN][kD];
  __shared__ float vs[kWN][kD];
  __shared__ float sc[kH * kWN * kWN];  // 648
  __shared__ float cs[kH * kWN];        // 72

  const int tid = threadIdx.x;
  const int g   = blockIdx.x;       // flat (b, n)
  const int b   = g >> 13;          // /kL
  const int n   = g & (kL - 1);
  const float scale = 0.17677669529663687f;  // 1/sqrt(32)

  const float* Q = QKV;
  const float* K = QKV + (size_t)kRows * kD;
  const float* V = QKV + (size_t)2 * kRows * kD;

  for (int w = 0; w < kWN; ++w) {
    int t = n + w - kP;
    bool in = (t >= 0) && (t < kL);
    size_t ro = (size_t)(b * kL + (in ? t : 0)) * kD + tid;
    float xq = in ? Q[ro] : 0.f;
    float xk = in ? K[ro] : 0.f;
    float xv = in ? V[ro] : 0.f;
    qs[w][tid] = xq + Ppos[(0 * kWN + w) * kD + tid];
    ks[w][tid] = xk + Ppos[(1 * kWN + w) * kD + tid];
    vs[w][tid] = xv + Ppos[(2 * kWN + w) * kD + tid];
  }
  __syncthreads();

  for (int idx = tid; idx < kH * kWN * kWN; idx += 256) {
    int h  = idx / 81;
    int r  = idx - h * 81;
    int qi = r / 9;
    int ki = r - qi * 9;
    const float* qp = &qs[qi][h * kHD];
    const float* kp = &ks[ki][h * kHD];
    float s = 0.f;
#pragma unroll
    for (int d = 0; d < kHD; ++d) s = fmaf(qp[d], kp[d], s);
    sc[idx] = s * scale;
  }
  __syncthreads();

  if (tid < kH * kWN) {  // softmax over k for each (h, qi)
    int h = tid / 9, qi = tid - h * 9;
    float* row = &sc[h * 81 + qi * 9];
    float m = row[0];
    for (int k = 1; k < 9; ++k) m = fmaxf(m, row[k]);
    float e[9];
    float s = 0.f;
    for (int k = 0; k < 9; ++k) { e[k] = __expf(row[k] - m); s += e[k]; }
    float inv = 1.f / s;
    for (int k = 0; k < 9; ++k) row[k] = e[k] * inv;
  }
  __syncthreads();

  if (tid < kH * kWN) {  // column sums over q for each (h, ki)
    int h = tid / 9, ki = tid - h * 9;
    float s = 0.f;
    for (int qi = 0; qi < 9; ++qi) s += sc[h * 81 + qi * 9 + ki];
    cs[h * 9 + ki] = s;
  }
  __syncthreads();

  {
    int h = tid >> 5;  // tid = h*32 + d
    float a = 0.f;
#pragma unroll
    for (int ki = 0; ki < 9; ++ki) a = fmaf(cs[h * 9 + ki], vs[ki][tid], a);
    ctx[(size_t)g * kD + tid] = a;
  }
}

// ---------------- fused output: LN(x + ctx@Wo) -> LN(a@Wout^T + b + a) -> relu ----------------
__global__ __launch_bounds__(128) void fused_out_kernel(
    const float* __restrict__ ctx, const float* __restrict__ x,
    const bf16_t* __restrict__ WoT, const bf16_t* __restrict__ WoutB,
    const float* __restrict__ b_out,
    const float* __restrict__ g1, const float* __restrict__ b1,
    const float* __restrict__ g2, const float* __restrict__ b2,
    float* __restrict__ out) {
  __shared__ float  ymat[16][kD];
  __shared__ float  zmat[16][kD];
  __shared__ bf16_t abmat[16][kD];
  __shared__ float  mu[16], rs[16];

  const int lane = threadIdx.x & 31;
  const int wave = threadIdx.x >> 5;
  const int tid  = threadIdx.x;
  const int row0 = blockIdx.x * 16;

  // pass 1: y = x + ctx @ W_o
  {
    v8f acc[4] = {};
#pragma unroll
    for (int kk = 0; kk < 8; ++kk) {
      int k0 = kk * 32;
      v16bf a = load_f32_tile_cvt(ctx, row0, k0, lane);
#pragma unroll
      for (int t = 0; t < 4; ++t) {
        v16bf b = load_bf16_tile(WoT, wave * 64 + t * 16, k0, lane);
        acc[t] = wmma_bf16(a, b, acc[t]);
      }
    }
    const int n  = lane & 15;
    const int mo = (lane >> 4) * 8;
#pragma unroll
    for (int t = 0; t < 4; ++t) {
      int col0 = wave * 64 + t * 16;
#pragma unroll
      for (int r = 0; r < 8; ++r) {
        int rr = mo + r;
        ymat[rr][col0 + n] = acc[t][r] + x[(size_t)(row0 + rr) * kD + col0 + n];
      }
    }
  }
  __syncthreads();

  if (tid < 16) {  // LN1 stats
    float s = 0.f, sq = 0.f;
    for (int c = 0; c < kD; ++c) { float v = ymat[tid][c]; s += v; sq = fmaf(v, v, sq); }
    float m = s * (1.f / kD);
    float var = sq * (1.f / kD) - m * m;
    mu[tid] = m;
    rs[tid] = rsqrtf(var + 1e-5f);
  }
  __syncthreads();

  for (int idx = tid; idx < 16 * kD; idx += 128) {
    int r = idx >> 8, c = idx & 255;
    float a = (ymat[r][c] - mu[r]) * rs[r] * g1[c] + b1[c];
    ymat[r][c]  = a;           // attn_res fp32 (for residual)
    abmat[r][c] = (bf16_t)a;   // attn_res bf16 (GEMM A operand)
  }
  __syncthreads();

  // pass 2: z = attn_res @ W_out^T + b_out + attn_res
  {
    v8f acc[4] = {};
    const int m  = lane & 15;
    const int ko = (lane >> 4) << 3;
#pragma unroll
    for (int kk = 0; kk < 8; ++kk) {
      int k0 = kk * 32;
      v16bf a;
#pragma unroll
      for (int i = 0; i < 8; ++i) {
        a[i]     = abmat[m][k0 + ko + i];
        a[i + 8] = abmat[m][k0 + ko + 16 + i];
      }
#pragma unroll
      for (int t = 0; t < 4; ++t) {
        v16bf b = load_bf16_tile(WoutB, wave * 64 + t * 16, k0, lane);
        acc[t] = wmma_bf16(a, b, acc[t]);
      }
    }
    const int n  = lane & 15;
    const int mo = (lane >> 4) * 8;
#pragma unroll
    for (int t = 0; t < 4; ++t) {
      int col0 = wave * 64 + t * 16;
#pragma unroll
      for (int r = 0; r < 8; ++r) {
        int rr = mo + r;
        int c  = col0 + n;
        zmat[rr][c] = acc[t][r] + b_out[c] + ymat[rr][c];
      }
    }
  }
  __syncthreads();

  if (tid < 16) {  // LN2 stats
    float s = 0.f, sq = 0.f;
    for (int c = 0; c < kD; ++c) { float v = zmat[tid][c]; s += v; sq = fmaf(v, v, sq); }
    float m = s * (1.f / kD);
    float var = sq * (1.f / kD) - m * m;
    mu[tid] = m;
    rs[tid] = rsqrtf(var + 1e-5f);
  }
  __syncthreads();

  for (int idx = tid; idx < 16 * kD; idx += 128) {
    int r = idx >> 8, c = idx & 255;
    float v = (zmat[r][c] - mu[r]) * rs[r] * g2[c] + b2[c];
    out[(size_t)(row0 + r) * kD + c] = v > 0.f ? v : 0.f;
  }
}

// ---------------- launch ----------------

extern "C" void kernel_launch(void* const* d_in, const int* in_sizes, int n_in,
                              void* d_out, int out_size, void* d_ws, size_t ws_size,
                              hipStream_t stream) {
  const float* x     = (const float*)d_in[0];
  const float* W_q   = (const float*)d_in[1];
  const float* W_k   = (const float*)d_in[2];
  const float* W_v   = (const float*)d_in[3];
  const float* W_o   = (const float*)d_in[4];
  const float* pos   = (const float*)d_in[5];
  const float* W_out = (const float*)d_in[6];
  const float* b_out = (const float*)d_in[7];
  const float* g1    = (const float*)d_in[8];
  const float* b1    = (const float*)d_in[9];
  const float* g2    = (const float*)d_in[10];
  const float* b2    = (const float*)d_in[11];
  float* out = (float*)d_out;

  char* ws = (char*)d_ws;
  size_t off = 0;
  bf16_t* xbf = (bf16_t*)(ws + off); off += (size_t)kRows * kD * sizeof(bf16_t);
  bf16_t* WB  = (bf16_t*)(ws + off); off += (size_t)5 * kD * kD * sizeof(bf16_t);
  float* Ppos = (float*)(ws + off);  off += (size_t)3 * kWN * kD * sizeof(float);
  off = (off + 255) & ~(size_t)255;
  float* QKV = (float*)(ws + off);   off += (size_t)3 * kRows * kD * sizeof(float);
  float* ctx = (float*)(ws + off);   off += (size_t)kRows * kD * sizeof(float);
  (void)ws_size; (void)in_sizes; (void)n_in; (void)out_size;

  cvt_x_kernel<<<(kRows * kD + 255) / 256, 256, 0, stream>>>(x, xbf, kRows * kD);
  prep_w_kernel<<<(5 * kD * kD) / 256, 256, 0, stream>>>(W_q, W_k, W_v, W_o, W_out, WB);
  posproj_kernel<<<3 * kWN, 256, 0, stream>>>(pos, W_q, W_k, W_v, Ppos);

  qkv_gemm_kernel<<<kRows / 16, 128, 0, stream>>>(xbf, WB, QKV);
  attn_kernel<<<kRows, 256, 0, stream>>>(QKV, Ppos, ctx);
  fused_out_kernel<<<kRows / 16, 128, 0, stream>>>(
      ctx, x, WB + (size_t)3 * kD * kD, WB + (size_t)4 * kD * kD,
      b_out, g1, b1, g2, b2, out);
}
